// Informer_49881750175912
// MI455X (gfx1250) — compile-verified, matrix operand order checked by reading, not run on hardware
//
#include <hip/hip_runtime.h>
#include <hip/hip_bf16.h>

typedef _Float16 v16h __attribute__((ext_vector_type(16)));
typedef _Float16 h8   __attribute__((ext_vector_type(8)));
typedef float    v8f  __attribute__((ext_vector_type(8)));

__device__ __forceinline__ float gelu_exact(float x) {
    return 0.5f * x * (1.0f + erff(x * 0.70710678118654752f));
}

__device__ __forceinline__ h8 cvt8(float4 a, float4 b) {
    h8 r;
    r[0] = (_Float16)a.x; r[1] = (_Float16)a.y; r[2] = (_Float16)a.z; r[3] = (_Float16)a.w;
    r[4] = (_Float16)b.x; r[5] = (_Float16)b.y; r[6] = (_Float16)b.z; r[7] = (_Float16)b.w;
    return r;
}

// load one wmma 16-half fragment: 8 contiguous halves at p, 8 at p+16
// (k-mapping per ISA 7.12.2: lane half h covers k = h*8..h*8+7 and 16+h*8..16+h*8+7)
__device__ __forceinline__ v16h ldfrag(const _Float16* p) {
    h8 lo = *(const h8*)p;
    h8 hi = *(const h8*)(p + 16);
    return __builtin_shufflevector(lo, hi, 0,1,2,3,4,5,6,7,8,9,10,11,12,13,14,15);
}

// ---------------------------------------------------------------------------
// WMMA GEMM: C[M,N] = act(A[M,K] @ W[K,N] + bias[N]) ; f32 in/out, f16 compute
// BM=128, BN=64, BK=32. 256 threads = 8 waves. Wave (wr=wave&3, wc=wave>>2)
// owns rows {wr*16, wr*16+64} x cols {wc*32, wc*32+16}: 4 wmma / wave / step.
// LDS: A row-major (m x k, pitch 40), B transposed (n x k, pitch 40) so both
// fragments are two ds_load_b128 per lane.
// ---------------------------------------------------------------------------
__global__ __launch_bounds__(256)
void gemm_f16_kernel(const float* __restrict__ A, const float* __restrict__ W,
                     const float* __restrict__ bias, float* __restrict__ C,
                     int M, int N, int K, int act)
{
    constexpr int BM = 128, BN = 64, BK = 32, PAD = 40;
    __shared__ _Float16 lA[BM * PAD];
    __shared__ _Float16 lB[BN * PAD];

    const int tid  = threadIdx.x;
    const int lane = tid & 31;
    const int wave = tid >> 5;
    const int wr   = wave & 3;
    const int wc   = wave >> 2;
    const int m0   = blockIdx.y * BM;
    const int n0   = blockIdx.x * BN;
    const int half = lane >> 4;
    const int l16  = lane & 15;

    v8f acc00 = {0,0,0,0,0,0,0,0};
    v8f acc01 = {0,0,0,0,0,0,0,0};
    v8f acc10 = {0,0,0,0,0,0,0,0};
    v8f acc11 = {0,0,0,0,0,0,0,0};

    const int  r0 = tid >> 1;            // A stage: row 0..127
    const int  cA = (tid & 1) * 16;      // A stage: col base 0/16
    const int  rk = tid >> 3;            // B stage: k row 0..31
    const int  nb = (tid & 7) * 8;       // B stage: n base 0..56
    const bool aligned = ((K & 3) == 0) && ((N & 3) == 0) &&
                         (m0 + BM <= M) && (n0 + BN <= N);

    for (int k0 = 0; k0 < K; k0 += BK) {
        if (aligned && (k0 + BK <= K)) {
            // ---- fast path: vectorized global loads, b128 LDS stores ----
            const float4* pa = (const float4*)(A + (size_t)(m0 + r0) * K + k0 + cA);
            float4 a0 = pa[0], a1 = pa[1], a2 = pa[2], a3 = pa[3];
            *(h8*)&lA[r0 * PAD + cA]     = cvt8(a0, a1);
            *(h8*)&lA[r0 * PAD + cA + 8] = cvt8(a2, a3);

            const float4* pw = (const float4*)(W + (size_t)(k0 + rk) * N + n0 + nb);
            float4 w0 = pw[0], w1 = pw[1];
            lB[(nb + 0) * PAD + rk] = (_Float16)w0.x;
            lB[(nb + 1) * PAD + rk] = (_Float16)w0.y;
            lB[(nb + 2) * PAD + rk] = (_Float16)w0.z;
            lB[(nb + 3) * PAD + rk] = (_Float16)w0.w;
            lB[(nb + 4) * PAD + rk] = (_Float16)w1.x;
            lB[(nb + 5) * PAD + rk] = (_Float16)w1.y;
            lB[(nb + 6) * PAD + rk] = (_Float16)w1.z;
            lB[(nb + 7) * PAD + rk] = (_Float16)w1.w;

            if (k0 + BK < K) {
                __builtin_prefetch(A + (size_t)(m0 + r0) * K + k0 + BK + cA, 0, 0);
                __builtin_prefetch(W + (size_t)(k0 + BK + rk) * N + n0 + nb, 0, 0);
            }
        } else {
            // ---- slow path: per-element guarded, zero-filled ----
            #pragma unroll
            for (int j = 0; j < 16; ++j) {
                int c = cA + j, gk = k0 + c;
                float v = ((m0 + r0) < M && gk < K) ? A[(size_t)(m0 + r0) * K + gk] : 0.0f;
                lA[r0 * PAD + c] = (_Float16)v;
            }
            int gk = k0 + rk;
            #pragma unroll
            for (int j = 0; j < 8; ++j) {
                int gn = n0 + nb + j;
                float v = (gk < K && gn < N) ? W[(size_t)gk * N + gn] : 0.0f;
                lB[(nb + j) * PAD + rk] = (_Float16)v;
            }
        }
        __syncthreads();

        const int ra0 = wr * 16 + l16;
        const int nb0 = wc * 32 + l16;
        v16h a0 = ldfrag(&lA[ra0 * PAD + half * 8]);
        v16h a1 = ldfrag(&lA[(ra0 + 64) * PAD + half * 8]);
        v16h b0 = ldfrag(&lB[nb0 * PAD + half * 8]);
        v16h b1 = ldfrag(&lB[(nb0 + 16) * PAD + half * 8]);

        acc00 = __builtin_amdgcn_wmma_f32_16x16x32_f16(false, a0, false, b0, (short)0, acc00, false, false);
        acc01 = __builtin_amdgcn_wmma_f32_16x16x32_f16(false, a0, false, b1, (short)0, acc01, false, false);
        acc10 = __builtin_amdgcn_wmma_f32_16x16x32_f16(false, a1, false, b0, (short)0, acc10, false, false);
        acc11 = __builtin_amdgcn_wmma_f32_16x16x32_f16(false, a1, false, b1, (short)0, acc11, false, false);
        __syncthreads();
    }

    // epilogue (C/D layout: lane<16 -> M=r, lane>=16 -> M=8+r; N = lane&15)
    #pragma unroll
    for (int rt = 0; rt < 2; ++rt) {
        #pragma unroll
        for (int r = 0; r < 8; ++r) {
            int row = m0 + wr * 16 + rt * 64 + half * 8 + r;
            if (row >= M) continue;
            int c0 = n0 + wc * 32 + l16;
            int c1 = c0 + 16;
            float v0 = (rt == 0) ? acc00[r] : acc10[r];
            float v1 = (rt == 0) ? acc01[r] : acc11[r];
            if (c0 < N) {
                float v = v0 + bias[c0];
                if (act == 1) v = gelu_exact(v);
                C[(size_t)row * N + c0] = v;
            }
            if (c1 < N) {
                float v = v1 + bias[c1];
                if (act == 1) v = gelu_exact(v);
                C[(size_t)row * N + c1] = v;
            }
        }
    }
}

// ---------------------------------------------------------------------------
// im2col for circular width-3 conv: Y[(b*L+l), t*C+c] = X[b, (l+t-1) mod L, c]
// ---------------------------------------------------------------------------
__global__ void im2col_circ_kernel(const float* __restrict__ X, float* __restrict__ Y,
                                   int B, int L, int Cin)
{
    long long t = (long long)blockIdx.x * blockDim.x + threadIdx.x;
    long long total = (long long)B * L * 3 * Cin;
    if (t >= total) return;
    int c   = (int)(t % Cin);
    int tap = (int)((t / Cin) % 3);
    int l   = (int)((t / ((long long)Cin * 3)) % L);
    int b   = (int)(t / ((long long)Cin * 3 * L));
    int ls  = (l + tap - 1 + L) % L;
    Y[t] = X[((size_t)(b * L + ls)) * Cin + c];
}

// per-channel batch-norm statistics over all rows
__global__ __launch_bounds__(256)
void bn_stats_kernel(const float* __restrict__ X, float* __restrict__ mean,
                     float* __restrict__ var, int Mrows)
{
    __shared__ float s1[256], s2[256];
    int c = blockIdx.x, t = threadIdx.x;
    float a = 0.f, b = 0.f;
    for (int r = t; r < Mrows; r += 256) {
        float v = X[(size_t)r * 512 + c];
        a += v; b += v * v;
    }
    s1[t] = a; s2[t] = b; __syncthreads();
    for (int s = 128; s > 0; s >>= 1) {
        if (t < s) { s1[t] += s1[t + s]; s2[t] += s2[t + s]; }
        __syncthreads();
    }
    if (t == 0) {
        float m = s1[0] / (float)Mrows;
        mean[c] = m;
        var[c]  = s2[0] / (float)Mrows - m * m;
    }
}

// mode 0: plain BN ; 1: BN + positional encoding ; 2: BN + ELU
__global__ void bn_apply_kernel(const float* __restrict__ X, float* __restrict__ Y,
                                const float* __restrict__ mean, const float* __restrict__ var,
                                const float* __restrict__ g, const float* __restrict__ b,
                                int Mrows, int L, int mode)
{
    long long t = (long long)blockIdx.x * blockDim.x + threadIdx.x;
    if (t >= (long long)Mrows * 512) return;
    int c = (int)(t % 512);
    int row = (int)(t / 512);
    float v = (X[t] - mean[c]) * rsqrtf(var[c] + 1e-5f) * g[c] + b[c];
    if (mode == 1) {
        int l = row % L;
        float div = __expf((float)(c & ~1) * (-9.210340371976184f / 512.0f));
        float ang = (float)l * div;
        v += (c & 1) ? __cosf(ang) : __sinf(ang);
    } else if (mode == 2) {
        v = v > 0.f ? v : (__expf(v) - 1.0f);
    }
    Y[t] = v;
}

// row LayerNorm over 512 cols, fused optional residual add (Y = LN(X[+R]))
__global__ __launch_bounds__(256)
void layernorm_kernel(const float* __restrict__ X, const float* __restrict__ R,
                      const float* __restrict__ g, const float* __restrict__ be,
                      float* __restrict__ Y)
{
    __shared__ float s1[256], s2[256];
    int row = blockIdx.x, t = threadIdx.x;
    const float* x = X + (size_t)row * 512;
    float a  = x[t]       + (R ? R[(size_t)row * 512 + t]       : 0.f);
    float b2 = x[t + 256] + (R ? R[(size_t)row * 512 + t + 256] : 0.f);
    s1[t] = a + b2; s2[t] = a * a + b2 * b2;
    __syncthreads();
    for (int s = 128; s > 0; s >>= 1) {
        if (t < s) { s1[t] += s1[t + s]; s2[t] += s2[t + s]; }
        __syncthreads();
    }
    float m = s1[0] / 512.0f;
    float v = s2[0] / 512.0f - m * m;
    float rinv = rsqrtf(v + 1e-5f);
    Y[(size_t)row * 512 + t]       = (a  - m) * rinv * g[t]       + be[t];
    Y[(size_t)row * 512 + t + 256] = (b2 - m) * rinv * g[t + 256] + be[t + 256];
}

// sparsity measure: M[b,h,q] = max_s(Q.K_s) - mean_s(Q.K_s) over U sampled keys
__global__ void attn_m_kernel(const float* __restrict__ Q, const float* __restrict__ Km,
                              float* __restrict__ Mm, int B, int Lq, int Lk, int U)
{
    int t = blockIdx.x * blockDim.x + threadIdx.x;
    int total = B * 8 * Lq;
    if (t >= total) return;
    int q = t % Lq;
    int h = (t / Lq) % 8;
    int b = t / (Lq * 8);
    const float* qp = Q + ((size_t)(b * Lq + q)) * 512 + h * 64;
    float qv[64];
    #pragma unroll
    for (int d = 0; d < 64; ++d) qv[d] = qp[d];
    float mx = -3.4e38f, sum = 0.f;
    unsigned seed = 0x9E3779B9u ^ (unsigned)t * 2654435761u;
    for (int s = 0; s < U; ++s) {
        seed = seed * 1664525u + 1013904223u;
        int k = (int)(seed % (unsigned)Lk);
        const float* kp = Km + ((size_t)(b * Lk + k)) * 512 + h * 64;
        float dot = 0.f;
        #pragma unroll
        for (int d = 0; d < 64; ++d) dot += qv[d] * kp[d];
        mx = fmaxf(mx, dot); sum += dot;
    }
    Mm[t] = mx - sum / (float)U;
}

// per (b,h): top-u indices of M
__global__ __launch_bounds__(256)
void topk_kernel(const float* __restrict__ Mm, int* __restrict__ Mtop, int Lq, int u)
{
    __shared__ float mbuf[2048];
    __shared__ float rv[256];
    __shared__ int   ri[256];
    int bh = blockIdx.x;
    const float* src = Mm + (size_t)bh * Lq;
    for (int i = threadIdx.x; i < Lq; i += 256) mbuf[i] = src[i];
    __syncthreads();
    for (int r = 0; r < u; ++r) {
        float best = -3.4e38f; int bi = 0;
        for (int i = threadIdx.x; i < Lq; i += 256)
            if (mbuf[i] > best) { best = mbuf[i]; bi = i; }
        rv[threadIdx.x] = best; ri[threadIdx.x] = bi;
        __syncthreads();
        for (int s = 128; s > 0; s >>= 1) {
            if (threadIdx.x < s && rv[threadIdx.x + s] > rv[threadIdx.x]) {
                rv[threadIdx.x] = rv[threadIdx.x + s];
                ri[threadIdx.x] = ri[threadIdx.x + s];
            }
            __syncthreads();
        }
        if (threadIdx.x == 0) {
            Mtop[(size_t)bh * u + r] = ri[0];
            mbuf[ri[0]] = -3.4e38f;
        }
        __syncthreads();
    }
}

__global__ void colmean_kernel(const float* __restrict__ V, float* __restrict__ vm,
                               int B, int Lk)
{
    int t = blockIdx.x * blockDim.x + threadIdx.x;
    if (t >= B * 512) return;
    int c = t % 512, b = t / 512;
    float s = 0.f;
    for (int l = 0; l < Lk; ++l) s += V[((size_t)(b * Lk + l)) * 512 + c];
    vm[t] = s / (float)Lk;
}

__global__ void ctx_bcast_kernel(const float* __restrict__ vm, float* __restrict__ ctx,
                                 int B, int Lq)
{
    long long t = (long long)blockIdx.x * blockDim.x + threadIdx.x;
    if (t >= (long long)B * Lq * 512) return;
    int c = (int)(t % 512);
    int b = (int)(t / ((long long)Lq * 512));
    ctx[t] = vm[b * 512 + c];
}

__global__ void ctx_cumsum_kernel(const float* __restrict__ V, float* __restrict__ ctx,
                                  int B, int L)
{
    int t = blockIdx.x * blockDim.x + threadIdx.x;
    if (t >= B * 512) return;
    int c = t % 512, b = t / 512;
    float s = 0.f;
    for (int l = 0; l < L; ++l) {
        s += V[((size_t)(b * L + l)) * 512 + c];
        ctx[((size_t)(b * L + l)) * 512 + c] = s;
    }
}

// full softmax-attention for the u selected queries; scatter into context rows
__global__ __launch_bounds__(256)
void attn_update_kernel(const float* __restrict__ Q, const float* __restrict__ Km,
                        const float* __restrict__ V, const int* __restrict__ Mtop,
                        float* __restrict__ ctx, int B, int Lq, int Lk, int u, int causal)
{
    __shared__ float sc[2048];
    __shared__ float red[256];
    __shared__ float qv[64];
    __shared__ float part[256];
    int blk = blockIdx.x;
    int ui = blk % u;
    int h  = (blk / u) % 8;
    int b  = blk / (u * 8);
    int qidx = Mtop[((size_t)(b * 8 + h)) * u + ui];
    int t = threadIdx.x;
    if (t < 64) qv[t] = Q[((size_t)(b * Lq + qidx)) * 512 + h * 64 + t];
    __syncthreads();
    int kmax = causal ? (qidx + 1) : Lk;
    float lmax = -3.4e38f;
    for (int k = t; k < kmax; k += 256) {
        const float* kp = Km + ((size_t)(b * Lk + k)) * 512 + h * 64;
        float dot = 0.f;
        #pragma unroll
        for (int d = 0; d < 64; ++d) dot += qv[d] * kp[d];
        dot *= 0.125f;                 // 1/sqrt(64)
        sc[k] = dot;
        lmax = fmaxf(lmax, dot);
    }
    red[t] = lmax; __syncthreads();
    for (int s = 128; s > 0; s >>= 1) { if (t < s) red[t] = fmaxf(red[t], red[t + s]); __syncthreads(); }
    float gmax = red[0]; __syncthreads();
    float lsum = 0.f;
    for (int k = t; k < kmax; k += 256) {
        float e = __expf(sc[k] - gmax);
        sc[k] = e; lsum += e;
    }
    red[t] = lsum; __syncthreads();
    for (int s = 128; s > 0; s >>= 1) { if (t < s) red[t] += red[t + s]; __syncthreads(); }
    float inv = 1.0f / red[0]; __syncthreads();
    int dd = t & 63, chunk = t >> 6;
    float acc = 0.f;
    for (int k = chunk; k < kmax; k += 4)
        acc += sc[k] * V[((size_t)(b * Lk + k)) * 512 + h * 64 + dd];
    part[t] = acc; __syncthreads();
    if (t < 64) {
        float v = (part[t] + part[t + 64] + part[t + 128] + part[t + 192]) * inv;
        ctx[((size_t)(b * Lq + qidx)) * 512 + h * 64 + dd] = v;
    }
}

// max-pool width 3, stride 2, pad 1 with -inf
__global__ void maxpool_kernel(const float* __restrict__ X, float* __restrict__ Y,
                               int B, int L)
{
    int Lout = L / 2;
    long long t = (long long)blockIdx.x * blockDim.x + threadIdx.x;
    if (t >= (long long)B * Lout * 512) return;
    int c  = (int)(t % 512);
    int lo = (int)((t / 512) % Lout);
    int b  = (int)(t / ((long long)512 * Lout));
    float m = -3.4e38f;
    #pragma unroll
    for (int tt = 0; tt < 3; ++tt) {
        int li = 2 * lo - 1 + tt;
        if (li >= 0 && li < L) m = fmaxf(m, X[((size_t)(b * L + li)) * 512 + c]);
    }
    Y[t] = m;
}

// d_out[b, t, c] = P[b*1024 + 512 + t, c], c<21, t<512
__global__ void slice_out_kernel(const float* __restrict__ P, float* __restrict__ O)
{
    int t = blockIdx.x * blockDim.x + threadIdx.x;
    if (t >= 4 * 512 * 21) return;
    int c  = t % 21;
    int tt = (t / 21) % 512;
    int b  = t / (21 * 512);
    O[t] = P[((size_t)(b * 1024) + 512 + tt) * 21 + c];
}

// ---------------------------------------------------------------------------
extern "C" void kernel_launch(void* const* d_in, const int* in_sizes, int n_in,
                              void* d_out, int out_size, void* d_ws, size_t ws_size,
                              hipStream_t stream)
{
    (void)in_sizes; (void)n_in; (void)out_size; (void)ws_size;
    const float* enc_inp = (const float*)d_in[0];
    const float* dec_inp = (const float*)d_in[1];

    int P = 2;
    auto nxt = [&]() { return (const float*)d_in[P++]; };

    const float* e_cw = nxt(); const float* e_cb = nxt();
    const float* e_bg = nxt(); const float* e_bb = nxt();
    const float* d_cw = nxt(); const float* d_cb = nxt();
    const float* d_bg = nxt(); const float* d_bb = nxt();

    struct AttnW { const float *qw,*qb,*kw,*kb,*vw,*vb,*ow,*ob; };
    auto rdAttn = [&]() {
        AttnW w;
        w.qw = nxt(); w.qb = nxt(); w.kw = nxt(); w.kb = nxt();
        w.vw = nxt(); w.vb = nxt(); w.ow = nxt(); w.ob = nxt();
        return w;
    };
    struct EncL { AttnW a; const float *f1w,*f1b,*f2w,*f2b,*l1g,*l1b,*l2g,*l2b; };
    EncL enc[2];
    for (int i = 0; i < 2; ++i) {
        enc[i].a = rdAttn();
        enc[i].f1w = nxt(); enc[i].f1b = nxt();
        enc[i].f2w = nxt(); enc[i].f2b = nxt();
        enc[i].l1g = nxt(); enc[i].l1b = nxt();
        enc[i].l2g = nxt(); enc[i].l2b = nxt();
    }
    const float* dcv_w = nxt(); const float* dcv_b = nxt();
    const float* dcv_g = nxt(); const float* dcv_bb = nxt();
    const float* eln_g = nxt(); const float* eln_b = nxt();
    AttnW dself  = rdAttn();
    AttnW dcross = rdAttn();
    const float* df1w = nxt(); const float* df1b = nxt();
    const float* df2w = nxt(); const float* df2b = nxt();
    const float* dl1g = nxt(); const float* dl1b = nxt();
    const float* dl2g = nxt(); const float* dl2b = nxt();
    const float* dl3g = nxt(); const float* dl3b = nxt();
    const float* dln_g = nxt(); const float* dln_b = nxt();
    const float* pj_w = nxt(); const float* pj_b = nxt();

    // workspace (bump allocator)
    float* ws = (float*)d_ws;
    size_t off = 0;
    auto alloc = [&](size_t n) { float* p = ws + off; off += (n + 63) & ~(size_t)63; return p; };
    const int B = 4;
    float* bufBig = alloc((size_t)8192 * 2048);  // FFN hidden / distil im2col
    float* colE   = alloc((size_t)8192 * 64);    // embedding im2col (K=63)
    float* Xe     = alloc((size_t)8192 * 512);   // encoder activations
    float* Qb     = alloc((size_t)8192 * 512);
    float* Kb     = alloc((size_t)8192 * 512);
    float* Vb     = alloc((size_t)8192 * 512);
    float* CTX    = alloc((size_t)8192 * 512);
    float* TMP    = alloc((size_t)8192 * 512);
    float* ENC    = alloc((size_t)4096 * 512);   // encoder output
    float* Xd     = alloc((size_t)4096 * 512);   // decoder activations
    float* Mme    = alloc((size_t)(4 * 8 * 2048));
    int*   Mtop   = (int*)alloc(4 * 8 * 64);
    float* vmean  = alloc(4 * 512);
    float* bnm    = alloc(512);
    float* bnv    = alloc(512);
    float* pout   = alloc((size_t)4096 * 32);

    auto gemm = [&](const float* A, const float* W, const float* bias, float* C,
                    int M, int N, int K, int act) {
        dim3 g((N + 63) / 64, (M + 127) / 128);
        gemm_f16_kernel<<<g, dim3(256), 0, stream>>>(A, W, bias, C, M, N, K, act);
    };

    auto attn = [&](const AttnW& w, const float* qin, const float* kvin,
                    int Lq, int Lk, int u, int U, int causal, float* outbuf) {
        int Mq = B * Lq, Mk = B * Lk;
        gemm(qin,  w.qw, w.qb, Qb, Mq, 512, 512, 0);
        gemm(kvin, w.kw, w.kb, Kb, Mk, 512, 512, 0);
        gemm(kvin, w.vw, w.vb, Vb, Mk, 512, 512, 0);
        int tot = B * 8 * Lq;
        attn_m_kernel<<<(tot + 255) / 256, 256, 0, stream>>>(Qb, Kb, Mme, B, Lq, Lk, U);
        topk_kernel<<<B * 8, 256, 0, stream>>>(Mme, Mtop, Lq, u);
        if (causal) {
            ctx_cumsum_kernel<<<(B * 512 + 255) / 256, 256, 0, stream>>>(Vb, CTX, B, Lk);
        } else {
            colmean_kernel<<<(B * 512 + 255) / 256, 256, 0, stream>>>(Vb, vmean, B, Lk);
            long long bt = (long long)B * Lq * 512;
            ctx_bcast_kernel<<<(unsigned)((bt + 255) / 256), 256, 0, stream>>>(vmean, CTX, B, Lq);
        }
        attn_update_kernel<<<B * 8 * u, 256, 0, stream>>>(Qb, Kb, Vb, Mtop, CTX, B, Lq, Lk, u, causal);
        gemm(CTX, w.ow, w.ob, outbuf, Mq, 512, 512, 0);
    };

    auto embed = [&](const float* xin, int L, const float* cw, const float* cb,
                     const float* bg, const float* bb, float* xout) {
        int Mrows = B * L, Kc = 3 * 21;
        long long tot = (long long)Mrows * Kc;
        im2col_circ_kernel<<<(unsigned)((tot + 255) / 256), 256, 0, stream>>>(xin, colE, B, L, 21);
        gemm(colE, cw, cb, TMP, Mrows, 512, Kc, 0);
        bn_stats_kernel<<<512, 256, 0, stream>>>(TMP, bnm, bnv, Mrows);
        long long t2 = (long long)Mrows * 512;
        bn_apply_kernel<<<(unsigned)((t2 + 255) / 256), 256, 0, stream>>>(TMP, xout, bnm, bnv, bg, bb, Mrows, L, 1);
    };

    // ---------------- encoder ----------------
    embed(enc_inp, 2048, e_cw, e_cb, e_bg, e_bb, Xe);
    int Lenc = 2048;
    for (int i = 0; i < 2; ++i) {
        int Mrows = B * Lenc;
        int u = (Lenc >= 2048) ? 40 : 35;   // FACTOR * ceil(ln(L))
        attn(enc[i].a, Xe, Xe, Lenc, Lenc, u, u, 0, TMP);
        layernorm_kernel<<<Mrows, 256, 0, stream>>>(Xe, TMP, enc[i].l1g, enc[i].l1b, Xe);
        gemm(Xe, enc[i].f1w, enc[i].f1b, bufBig, Mrows, 2048, 512, 1);
        gemm(bufBig, enc[i].f2w, enc[i].f2b, TMP, Mrows, 512, 2048, 0);
        layernorm_kernel<<<Mrows, 256, 0, stream>>>(Xe, TMP, enc[i].l2g, enc[i].l2b, Xe);
        if (i == 0) {
            // distil: circular conv 512->512 (im2col GEMM, K=1536) + BN + ELU + maxpool/2
            long long tot = (long long)Mrows * 1536;
            im2col_circ_kernel<<<(unsigned)((tot + 255) / 256), 256, 0, stream>>>(Xe, bufBig, B, Lenc, 512);
            gemm(bufBig, dcv_w, dcv_b, TMP, Mrows, 512, 1536, 0);
            bn_stats_kernel<<<512, 256, 0, stream>>>(TMP, bnm, bnv, Mrows);
            long long t2 = (long long)Mrows * 512;
            bn_apply_kernel<<<(unsigned)((t2 + 255) / 256), 256, 0, stream>>>(TMP, TMP, bnm, bnv, dcv_g, dcv_bb, Mrows, Lenc, 2);
            long long t3 = (long long)B * (Lenc / 2) * 512;
            maxpool_kernel<<<(unsigned)((t3 + 255) / 256), 256, 0, stream>>>(TMP, Xe, B, Lenc);
            Lenc /= 2;
        }
    }
    layernorm_kernel<<<B * Lenc, 256, 0, stream>>>(Xe, nullptr, eln_g, eln_b, ENC);

    // ---------------- decoder ----------------
    embed(dec_inp, 1024, d_cw, d_cb, d_bg, d_bb, Xd);
    int Ld = 1024, Md = B * Ld, Lc = Lenc; // Lc = 1024
    attn(dself, Xd, Xd, Ld, Ld, 35, 35, 1, TMP);
    layernorm_kernel<<<Md, 256, 0, stream>>>(Xd, TMP, dl1g, dl1b, Xd);
    attn(dcross, Xd, ENC, Ld, Lc, 35, 35, 0, TMP);
    layernorm_kernel<<<Md, 256, 0, stream>>>(Xd, TMP, dl2g, dl2b, Xd);
    gemm(Xd, df1w, df1b, bufBig, Md, 2048, 512, 1);
    gemm(bufBig, df2w, df2b, TMP, Md, 512, 2048, 0);
    layernorm_kernel<<<Md, 256, 0, stream>>>(Xd, TMP, dl3g, dl3b, Xd);
    layernorm_kernel<<<Md, 256, 0, stream>>>(Xd, nullptr, dln_g, dln_b, Xd);
    gemm(Xd, pj_w, pj_b, pout, Md, 21, 512, 0);
    slice_out_kernel<<<(4 * 512 * 21 + 255) / 256, 256, 0, stream>>>(pout, (float*)d_out);
}
